// ClsGraphConv_84696755077584
// MI455X (gfx1250) — compile-verified, matrix operand order checked by reading, not run on hardware
//
#include <hip/hip_runtime.h>
#include <hip/hip_bf16.h>
#include <math.h>

typedef __attribute__((ext_vector_type(16))) _Float16 v16h;
typedef __attribute__((ext_vector_type(8)))  float    v8f;

#define D 64
#define N_REL 16
#define N_FAC 4
#define N_CLS 8
#define EPSN 1e-12f
#define TMP_T 0.2f

// ---------------------------------------------------------------- utility
__global__ void zero_kernel(float* __restrict__ p, long long n) {
    long long i = (long long)blockIdx.x * blockDim.x + threadIdx.x;
    long long stride = (long long)gridDim.x * blockDim.x;
    for (; i < n; i += stride) p[i] = 0.0f;
}

__global__ void init_out_kernel(const float* __restrict__ ent,
                                const float* __restrict__ usr,
                                float* __restrict__ out,
                                long long ne, long long ntot) {
    long long i = (long long)blockIdx.x * blockDim.x + threadIdx.x;
    long long stride = (long long)gridDim.x * blockDim.x;
    for (; i < ntot; i += stride)
        out[i] = (i < ne) ? ent[i] : usr[i - ne];
}

// ---------------------------------------------------------------- disen weight
// dw[f][col] = sum_r softmax(dwa[f])[r] * rel[r][col] ; 4x64 output, one block of 256
__global__ void disen_weight_kernel(const float* __restrict__ dwa,
                                    const float* __restrict__ rel,
                                    float* __restrict__ dw) {
    int t = threadIdx.x;            // 0..255
    int f = t >> 6;                 // 0..3
    int col = t & 63;
    float mx = -3.4e38f;
    for (int r = 0; r < N_REL; ++r) mx = fmaxf(mx, dwa[f * N_REL + r]);
    float w[N_REL]; float s = 0.0f;
    for (int r = 0; r < N_REL; ++r) { w[r] = __expf(dwa[f * N_REL + r] - mx); s += w[r]; }
    float inv = 1.0f / s;
    float acc = 0.0f;
    for (int r = 0; r < N_REL; ++r) acc += (w[r] * inv) * rel[r * D + col];
    dw[f * D + col] = acc;
}

// ---------------------------------------------------------------- attention (WMMA)
// S(16x16) = U_tile(16x64,f16) x W^T(64x16,f16), W rows: 0-7 cls_emb, 8-11 latent_emb,
// 12-15 zero. Two chained v_wmma_f32_16x16x32_f16 (K = 2*32).
// Softmax via wave-local LDS transpose: lane halves then process whole rows serially
// (lanes 0-15: class softmax over 8 cols; lanes 16-31: latent softmax over 4 cols),
// with contiguous b128 loads/stores instead of ds_bpermute butterflies.
__global__ void att_kernel(const float* __restrict__ u_cur,
                           const float* __restrict__ cls_emb,
                           const float* __restrict__ latent_emb,
                           float* __restrict__ att_cls,
                           float* __restrict__ att_fac,
                           int n_users) {
    const int lane = threadIdx.x & 31;
    const int waveInBlock = threadIdx.x >> 5;
    const int wavesPerBlock = blockDim.x >> 5;
    const int globalWave = blockIdx.x * wavesPerBlock + waveInBlock;
    const int totalWaves = gridDim.x * wavesPerBlock;

    const int h = lane >> 4;   // wave half
    const int m = lane & 15;   // A row index (user) and B col index (class/factor)

    // per-wave score tile; stride 20 floats (80 B): rows 16B-aligned, conflict-free
    __shared__ float stile[8][16 * 20];
    float* swave = stile[waveInBlock];

    // Stationary B operands (per ISA 16-bit operand layout, lane=N, halves split K)
    v16h b0, b1;
    #pragma unroll
    for (int e = 0; e < 16; ++e) {
        int k0 = 8 * h + (e & 7) + ((e & 8) ? 16 : 0);
        float w0, w1;
        if (m < 8)       { w0 = cls_emb[m * D + k0];          w1 = cls_emb[m * D + 32 + k0]; }
        else if (m < 12) { w0 = latent_emb[(m - 8) * D + k0]; w1 = latent_emb[(m - 8) * D + 32 + k0]; }
        else             { w0 = 0.0f;                         w1 = 0.0f; }
        b0[e] = (_Float16)w0;
        b1[e] = (_Float16)w1;
    }

    const int ntiles = n_users >> 4;   // n_users = 50000 -> 3125, exact
    for (int tile = globalWave; tile < ntiles; tile += totalWaves) {
        const int u0 = tile << 4;
        const float* up = u_cur + (size_t)(u0 + m) * D;
        v16h a0, a1;
        #pragma unroll
        for (int e = 0; e < 16; ++e) {
            int k0 = 8 * h + (e & 7) + ((e & 8) ? 16 : 0);
            a0[e] = (_Float16)up[k0];
            a1[e] = (_Float16)up[32 + k0];
        }
        v8f c = {};
        c = __builtin_amdgcn_wmma_f32_16x16x32_f16(false, a0, false, b0, (short)0, c, false, false);
        c = __builtin_amdgcn_wmma_f32_16x16x32_f16(false, a1, false, b1, (short)0, c, false, false);

        // scatter C tile to LDS: lane holds S[r + 8h][m] in c[r]
        #pragma unroll
        for (int r = 0; r < 8; ++r)
            swave[(r + 8 * h) * 20 + m] = c[r];

        // DS ops from the same wave are processed in order; keep the compiler honest:
        __builtin_amdgcn_wave_barrier();
        asm volatile("s_wait_dscnt 0x0" ::: "memory");

        const int row = m;                 // user row this lane finalizes
        const float* lr = swave + row * 20;
        if (h == 0) {
            // class softmax over cols 0..7 -> att_cls[u0+row][0..7]
            float4 s0 = *(const float4*)(lr);
            float4 s1 = *(const float4*)(lr + 4);
            float mx = fmaxf(fmaxf(fmaxf(s0.x, s0.y), fmaxf(s0.z, s0.w)),
                             fmaxf(fmaxf(s1.x, s1.y), fmaxf(s1.z, s1.w)));
            float4 e0, e1;
            e0.x = __expf(s0.x - mx); e0.y = __expf(s0.y - mx);
            e0.z = __expf(s0.z - mx); e0.w = __expf(s0.w - mx);
            e1.x = __expf(s1.x - mx); e1.y = __expf(s1.y - mx);
            e1.z = __expf(s1.z - mx); e1.w = __expf(s1.w - mx);
            float inv = 1.0f / (e0.x + e0.y + e0.z + e0.w + e1.x + e1.y + e1.z + e1.w);
            e0.x *= inv; e0.y *= inv; e0.z *= inv; e0.w *= inv;
            e1.x *= inv; e1.y *= inv; e1.z *= inv; e1.w *= inv;
            float* dst = att_cls + (size_t)(u0 + row) * N_CLS;
            *(float4*)(dst)     = e0;
            *(float4*)(dst + 4) = e1;
        } else {
            // latent softmax over cols 8..11 -> att_fac[u0+row][0..3]
            float4 s = *(const float4*)(lr + 8);
            float mx = fmaxf(fmaxf(s.x, s.y), fmaxf(s.z, s.w));
            float4 e;
            e.x = __expf(s.x - mx); e.y = __expf(s.y - mx);
            e.z = __expf(s.z - mx); e.w = __expf(s.w - mx);
            float inv = 1.0f / (e.x + e.y + e.z + e.w);
            e.x *= inv; e.y *= inv; e.z *= inv; e.w *= inv;
            *(float4*)(att_fac + (size_t)(u0 + row) * N_FAC) = e;
        }
        // next iteration overwrites the tile only after this wave's loads completed
        __builtin_amdgcn_wave_barrier();
    }
}

// ---------------------------------------------------------------- edge scatter
// 16 threads per edge, float4 per thread. Relation table cached in LDS (4 KB).
__global__ void edge_kernel(const float* __restrict__ e_cur,
                            const float* __restrict__ rel_emb,
                            const int* __restrict__ edge_index,
                            const int* __restrict__ edge_type,
                            float* __restrict__ ent_sum,
                            float* __restrict__ cnt,
                            int n_edges) {
    __shared__ float srel[N_REL * D];
    for (int i = threadIdx.x; i < N_REL * D; i += blockDim.x) srel[i] = rel_emb[i];
    __syncthreads();

    long long gid = (long long)blockIdx.x * blockDim.x + threadIdx.x;
    long long eidx = gid >> 4;
    int sub = (int)(gid & 15);
    if (eidx >= n_edges) return;

    const int head = edge_index[eidx];
    const int tail = edge_index[(size_t)n_edges + eidx];
    const int rt   = edge_type[eidx];

    const float4 ev = *(const float4*)(e_cur + (size_t)tail * D + sub * 4);
    const float4 rv = *(const float4*)(srel + rt * D + sub * 4);
    float* dst = ent_sum + (size_t)head * D + sub * 4;
    atomicAdd(dst + 0, ev.x * rv.x);
    atomicAdd(dst + 1, ev.y * rv.y);
    atomicAdd(dst + 2, ev.z * rv.z);
    atomicAdd(dst + 3, ev.w * rv.w);
    if (sub == 0) atomicAdd(cnt + head, 1.0f);
}

// ---------------------------------------------------------------- interaction scatter
// Fuses segment_sum(u_msg) and einsum('cud,uc->ud'): coef = val * (1 + att_cls[u,c])
__global__ void inter_kernel(const float* __restrict__ e_cur,
                             const float* __restrict__ att_cls,
                             const int* __restrict__ inter_user,
                             const int* __restrict__ inter_item,
                             const float* __restrict__ inter_val,
                             const int* __restrict__ inter_cls,
                             float* __restrict__ user_agg,
                             int n_inter) {
    long long gid = (long long)blockIdx.x * blockDim.x + threadIdx.x;
    long long i = gid >> 4;
    int sub = (int)(gid & 15);
    if (i >= n_inter) return;

    const int u  = inter_user[i];
    const int it = inter_item[i];
    const int c  = inter_cls[i];
    const float coef = inter_val[i] * (1.0f + att_cls[(size_t)u * N_CLS + c]);

    const float4 iv = *(const float4*)(e_cur + (size_t)it * D + sub * 4);
    float* dst = user_agg + (size_t)u * D + sub * 4;
    atomicAdd(dst + 0, coef * iv.x);
    atomicAdd(dst + 1, coef * iv.y);
    atomicAdd(dst + 2, coef * iv.z);
    atomicAdd(dst + 3, coef * iv.w);
}

// ---------------------------------------------------------------- entity finalize
// agg = ent_sum / max(cnt,1); e_next = l2norm(agg); out += e_next. One wave per row.
__global__ void entity_fin_kernel(const float* __restrict__ ent_sum,
                                  const float* __restrict__ cnt,
                                  float* __restrict__ e_next,
                                  float* __restrict__ out_ent,
                                  int n_entities) {
    const int lane = threadIdx.x & 31;
    const int row = (int)(((long long)blockIdx.x * blockDim.x + threadIdx.x) >> 5);
    if (row >= n_entities) return;

    const float inv = 1.0f / fmaxf(cnt[row], 1.0f);
    float2 v = *(const float2*)(ent_sum + (size_t)row * D + lane * 2);
    v.x *= inv; v.y *= inv;
    float ss = v.x * v.x + v.y * v.y;
    #pragma unroll
    for (int msk = 1; msk < 32; msk <<= 1) ss += __shfl_xor(ss, msk, 32);
    const float scale = 1.0f / fmaxf(sqrtf(ss), EPSN);
    v.x *= scale; v.y *= scale;
    *(float2*)(e_next + (size_t)row * D + lane * 2) = v;
    float2 o = *(float2*)(out_ent + (size_t)row * D + lane * 2);
    o.x += v.x; o.y += v.y;
    *(float2*)(out_ent + (size_t)row * D + lane * 2) = o;
}

// ---------------------------------------------------------------- user finalize
// user_agg += att_fac[u] @ disen_weight; u_next = l2norm(user_agg); out += u_next.
__global__ void user_fin_kernel(const float* __restrict__ user_agg,
                                const float* __restrict__ att_fac,
                                const float* __restrict__ dw,
                                float* __restrict__ u_next,
                                float* __restrict__ out_usr,
                                int n_users) {
    const int lane = threadIdx.x & 31;
    const int row = (int)(((long long)blockIdx.x * blockDim.x + threadIdx.x) >> 5);
    if (row >= n_users) return;

    const int col = lane * 2;
    float2 v = *(const float2*)(user_agg + (size_t)row * D + col);
    const float a0 = att_fac[(size_t)row * N_FAC + 0];
    const float a1 = att_fac[(size_t)row * N_FAC + 1];
    const float a2 = att_fac[(size_t)row * N_FAC + 2];
    const float a3 = att_fac[(size_t)row * N_FAC + 3];
    v.x += a0 * dw[0 * D + col]     + a1 * dw[1 * D + col]     + a2 * dw[2 * D + col]     + a3 * dw[3 * D + col];
    v.y += a0 * dw[0 * D + col + 1] + a1 * dw[1 * D + col + 1] + a2 * dw[2 * D + col + 1] + a3 * dw[3 * D + col + 1];
    float ss = v.x * v.x + v.y * v.y;
    #pragma unroll
    for (int msk = 1; msk < 32; msk <<= 1) ss += __shfl_xor(ss, msk, 32);
    const float scale = 1.0f / fmaxf(sqrtf(ss), EPSN);
    v.x *= scale; v.y *= scale;
    *(float2*)(u_next + (size_t)row * D + col) = v;
    float2 o = *(float2*)(out_usr + (size_t)row * D + col);
    o.x += v.x; o.y += v.y;
    *(float2*)(out_usr + (size_t)row * D + col) = o;
}

// ---------------------------------------------------------------- correlation loss (4x16, trivial)
__global__ void cor_kernel(const float* __restrict__ dwa, float* __restrict__ out) {
    float nt[N_FAC][N_REL];
    for (int i = 0; i < N_FAC; ++i) {
        float ss = 0.0f;
        for (int j = 0; j < N_REL; ++j) { float x = dwa[i * N_REL + j]; ss += x * x; }
        float n = fmaxf(sqrtf(ss), EPSN);
        for (int j = 0; j < N_REL; ++j) nt[i][j] = dwa[i * N_REL + j] / n;
    }
    float loss = 0.0f;
    for (int i = 0; i < N_FAC; ++i) {
        float rowsum = 0.0f, diag = 0.0f;
        for (int j = 0; j < N_FAC; ++j) {
            float d = 0.0f;
            for (int k = 0; k < N_REL; ++k) d += nt[i][k] * nt[j][k];
            float sc = expf(d / TMP_T);
            rowsum += sc;
            if (i == j) diag = sc;
        }
        loss -= logf(diag / rowsum);
    }
    *out = loss;
}

// ================================================================ launch
extern "C" void kernel_launch(void* const* d_in, const int* in_sizes, int n_in,
                              void* d_out, int out_size, void* d_ws, size_t ws_size,
                              hipStream_t stream) {
    const float* user_emb    = (const float*)d_in[0];
    const float* entity_emb  = (const float*)d_in[1];
    const float* cls_emb     = (const float*)d_in[2];
    const float* latent_emb  = (const float*)d_in[3];
    const float* rel_emb     = (const float*)d_in[4];
    const float* dwa         = (const float*)d_in[5];
    const int*   edge_index  = (const int*)d_in[6];
    const int*   edge_type   = (const int*)d_in[7];
    const int*   inter_user  = (const int*)d_in[8];
    const int*   inter_item  = (const int*)d_in[9];
    const float* inter_val   = (const float*)d_in[10];
    const int*   inter_cls   = (const int*)d_in[11];

    const int n_users    = in_sizes[0] / D;   // 50000
    const int n_entities = in_sizes[1] / D;   // 100000
    const int n_edges    = in_sizes[7];       // 1000000
    const int n_inter    = in_sizes[8];       // 500000

    // ---- workspace layout (floats)
    float* ws = (float*)d_ws;
    float* eA       = ws;                                  // n_entities*64
    float* uA       = eA + (size_t)n_entities * D;         // n_users*64
    float* ent_sum  = uA + (size_t)n_users * D;            // n_entities*64
    float* user_agg = ent_sum + (size_t)n_entities * D;    // n_users*64
    float* cnt      = user_agg + (size_t)n_users * D;      // n_entities
    float* att_cls  = cnt + (size_t)n_entities;            // n_users*8
    float* att_fac  = att_cls + (size_t)n_users * N_CLS;   // n_users*4
    float* dwbuf    = att_fac + (size_t)n_users * N_FAC;   // 4*64

    const long long zero_n = (long long)n_entities * D + (long long)n_users * D + n_entities;

    float* out_ent = (float*)d_out;
    float* out_usr = out_ent + (size_t)n_entities * D;
    float* out_cor = out_usr + (size_t)n_users * D;

    const long long ntot = (long long)n_entities * D + (long long)n_users * D;

    // output init: res = input embeddings
    init_out_kernel<<<1024, 256, 0, stream>>>(entity_emb, user_emb, out_ent,
                                              (long long)n_entities * D, ntot);
    // constant across hops
    disen_weight_kernel<<<1, 256, 0, stream>>>(dwa, rel_emb, dwbuf);

    for (int hop = 0; hop < 2; ++hop) {
        const float* e_cur = (hop == 0) ? entity_emb : eA;
        const float* u_cur = (hop == 0) ? user_emb : uA;

        // zero accumulators (ent_sum | user_agg | cnt are contiguous)
        zero_kernel<<<2048, 256, 0, stream>>>(ent_sum, zero_n);

        // WMMA attention scores + softmaxes
        att_kernel<<<128, 256, 0, stream>>>(u_cur, cls_emb, latent_emb,
                                            att_cls, att_fac, n_users);

        // KG edge aggregation
        {
            long long threads = (long long)n_edges * 16;
            int blocks = (int)((threads + 255) / 256);
            edge_kernel<<<blocks, 256, 0, stream>>>(e_cur, rel_emb, edge_index, edge_type,
                                                    ent_sum, cnt, n_edges);
        }
        // interaction aggregation (fused plain + class-weighted einsum)
        {
            long long threads = (long long)n_inter * 16;
            int blocks = (int)((threads + 255) / 256);
            inter_kernel<<<blocks, 256, 0, stream>>>(e_cur, att_cls, inter_user, inter_item,
                                                     inter_val, inter_cls, user_agg, n_inter);
        }
        // finalize (mean, +disen term, l2norm, accumulate into output)
        {
            int blocks = (int)(((long long)n_entities * 32 + 255) / 256);
            entity_fin_kernel<<<blocks, 256, 0, stream>>>(ent_sum, cnt, eA, out_ent, n_entities);
        }
        {
            int blocks = (int)(((long long)n_users * 32 + 255) / 256);
            user_fin_kernel<<<blocks, 256, 0, stream>>>(user_agg, att_fac, dwbuf, uA, out_usr, n_users);
        }
    }

    cor_kernel<<<1, 1, 0, stream>>>(dwa, out_cor);
    (void)n_in; (void)out_size; (void)ws_size;
}
